// DetectNet_90391881711707
// MI455X (gfx1250) — matrix operand order repaired and checked by reading, not compiled, not run on hardware
//
#include <hip/hip_runtime.h>
#include <hip/hip_bf16.h>

// ---------------------------------------------------------------------------
// Types for CDNA5 WMMA
// ---------------------------------------------------------------------------
typedef __attribute__((ext_vector_type(16))) __bf16          v16bf;
typedef __attribute__((ext_vector_type(16))) unsigned short  v16us;
typedef __attribute__((ext_vector_type(8)))  float           v8f;
typedef __attribute__((ext_vector_type(8)))  unsigned int    v8u;
typedef __attribute__((ext_vector_type(4)))  unsigned int    v4u;

// ---------------------------------------------------------------------------
// Network dimensions
// ---------------------------------------------------------------------------
#define HIN   640
#define WIN   960
#define CIN0  3

#define H1    313
#define W1    473
#define C1    32

#define HP1   156
#define WP1   236

#define H2    39
#define W2    59
#define C2    64

#define HP2   19
#define WP2   29

#define H3    4
#define W3    7
#define C3    128

#define FLATK 3584           // 4*7*128
#define DOUT  38400          // 40*60*8*2

#define NBOX  4800
#define VECL  8
#define NCLS  3
#define NPAD  8192           // NBOX padded to pow2 for bitonic sort

// Use the gfx1250 async global->LDS copy engine (ASYNCcnt-tracked) for the
// cooperative B-fragment staging. Inline asm keeps it portable across both
// toolchains (bypasses unprobed builtin signatures).
#define USE_ASYNC_LDS 1

// ---------------------------------------------------------------------------
// fp32 -> bf16 (round to nearest even)
// ---------------------------------------------------------------------------
static __device__ __forceinline__ unsigned short f32_to_bf16(float f) {
    unsigned int u = __builtin_bit_cast(unsigned int, f);
    unsigned int rounding = 0x7FFFu + ((u >> 16) & 1u);
    return (unsigned short)((u + rounding) >> 16);
}

// ---------------------------------------------------------------------------
// Pack a K x N row-major fp32 weight matrix into CDNA5 bf16 B-fragments.
// Fragment layout (per 32x16 K x N tile, wave32):
//   lanes 0-15 : column N = lane,      K = ktile*32 + {0..15}
//   lanes 16-31: column N = lane - 16, K = ktile*32 + {16..31}
//   dword d holds bf16 pair (K = base+2d, base+2d+1)
// Out layout: frag = kt*ntiles + nt ; out[(frag*32 + lane)*8 + d]
// ---------------------------------------------------------------------------
__global__ void pack_b_bf16(const float* __restrict__ W,
                            unsigned int* __restrict__ out,
                            int K, int N) {
    int frag = blockIdx.x;
    int lane = threadIdx.x;            // 0..31
    int ntiles = N >> 4;
    int kt = frag / ntiles;
    int nt = frag - kt * ntiles;
    int col   = nt * 16 + (lane & 15);
    int kbase = kt * 32 + ((lane & 16) ? 16 : 0);
    (void)K;
#pragma unroll
    for (int d = 0; d < 8; ++d) {
        unsigned int lo = f32_to_bf16(W[(kbase + 2 * d    ) * N + col]);
        unsigned int hi = f32_to_bf16(W[(kbase + 2 * d + 1) * N + col]);
        out[(frag * 32 + lane) * 8 + d] = lo | (hi << 16);
    }
}

// ---------------------------------------------------------------------------
// Implicit-GEMM convolution via v_wmma_f32_16x16x32_bf16.
//   M = Hout*Wout,  K = KH*KW*Cin,  N = Cout = NTILES*16
// Multiple waves per block; the full packed-B matrix is staged once per block
// into LDS (async global->LDS on gfx1250), then every wave's K-loop reads B
// fragments from LDS (ds_load_b128) while gathering A im2col-on-the-fly.
// ---------------------------------------------------------------------------
template <int NTILES, int CINT, int KWT>
__global__ void conv_wmma_bf16(const float* __restrict__ in,
                               const unsigned int* __restrict__ Bpack,
                               const float* __restrict__ bias,
                               float* __restrict__ out,
                               int M, int Wout, int Win, int stride, int ktiles) {
    extern __shared__ unsigned int Blds[];
    const int tid  = threadIdx.x;
    const int lane = tid & 31;
    const int wave = tid >> 5;
    const int wpb  = blockDim.x >> 5;
    const int Cout = NTILES * 16;
    const int Bwords = ktiles * NTILES * 32 * 8;   // dwords of packed B

    // ---- cooperative stage: packed B -> LDS (16 B per lane per issue) -----
#if USE_ASYNC_LDS
    for (int w = tid * 4; w < Bwords; w += blockDim.x * 4) {
        unsigned int       lds_off = (unsigned int)(size_t)(Blds + w);
        unsigned long long gaddr   = (unsigned long long)(size_t)(Bpack + w);
        asm volatile("global_load_async_to_lds_b128 %0, %1, off"
                     :: "v"(lds_off), "v"(gaddr) : "memory");
    }
    asm volatile("s_wait_asynccnt 0" ::: "memory");
#else
    for (int w = tid * 4; w < Bwords; w += blockDim.x * 4)
        *(v4u*)(Blds + w) = *(const v4u*)(Bpack + w);
#endif
    __syncthreads();

    const int mtile = blockIdx.x * wpb + wave;
    if (mtile * 16 >= M) return;                   // wave-uniform exit

    int row = mtile * 16 + (lane & 15);
    bool rowvalid = (row < M);
    int rc = rowvalid ? row : (M - 1);
    int oy = rc / Wout;
    int ox = rc - oy * Wout;
    int y0 = oy * stride;
    int x0 = ox * stride;
    const int khalf = (lane & 16) ? 16 : 0;

    v8f acc[NTILES];
#pragma unroll
    for (int nt = 0; nt < NTILES; ++nt)
        acc[nt] = (v8f){0.f, 0.f, 0.f, 0.f, 0.f, 0.f, 0.f, 0.f};

    for (int kt = 0; kt < ktiles; ++kt) {
        // ---- gather A fragment (16 bf16 per lane, im2col) -----------------
        v16us araw;
#pragma unroll
        for (int i = 0; i < 16; ++i) {
            int k  = kt * 32 + i + khalf;
            int ci = k % CINT;
            int t  = k / CINT;
            int kx = t % KWT;
            int ky = t / KWT;
            float v = rowvalid ? in[((y0 + ky) * Win + (x0 + kx)) * CINT + ci] : 0.0f;
            araw[i] = f32_to_bf16(v);
        }
        v16bf a = __builtin_bit_cast(v16bf, araw);

        // ---- NTILES WMMAs, B fragments from LDS ---------------------------
#pragma unroll
        for (int nt = 0; nt < NTILES; ++nt) {
            const unsigned int* bp = Blds + ((kt * NTILES + nt) * 32 + lane) * 8;
            v8u braw = *(const v8u*)bp;            // 2x ds_load_b128
            v16bf b  = __builtin_bit_cast(v16bf, braw);
            acc[nt]  = __builtin_amdgcn_wmma_f32_16x16x32_bf16(
                           false, a, false, b, (short)0, acc[nt], false, false);
        }
    }

    // ---- bias + ReLU + store (C/D layout: VGPR r -> M=r(+8), N=lane&15) ---
#pragma unroll
    for (int nt = 0; nt < NTILES; ++nt) {
        int col  = nt * 16 + (lane & 15);
        float bv = bias[col];
#pragma unroll
        for (int r = 0; r < 8; ++r) {
            int m = mtile * 16 + r + ((lane & 16) ? 8 : 0);
            if (m < M) {
                float v = acc[nt][r] + bv;
                out[m * Cout + col] = v > 0.f ? v : 0.f;
            }
        }
    }
}

// ---------------------------------------------------------------------------
// 2x2 max-pool, stride 2, VALID
// ---------------------------------------------------------------------------
__global__ void maxpool2(const float* __restrict__ in, float* __restrict__ out,
                         int Win, int C, int Hout, int Wout) {
    int idx = blockIdx.x * blockDim.x + threadIdx.x;
    int total = Hout * Wout * C;
    if (idx >= total) return;
    int c = idx % C;
    int t = idx / C;
    int x = t % Wout;
    int y = t / Wout;
    const float* p = in + ((2 * y) * Win + 2 * x) * C + c;
    float m = fmaxf(fmaxf(p[0], p[C]), fmaxf(p[Win * C], p[Win * C + C]));
    out[idx] = m;
}

// ---------------------------------------------------------------------------
// Dense GEMV: out[n] = relu(sum_k x[k]*W[k,n] + b[n]).  HBM-bound on W
// (550 MB fp32 ~ 24 us @ 23.3 TB/s) -> coalesced fp32 streaming is optimal.
// x staged in LDS; prefetch ahead on the weight stream.
// ---------------------------------------------------------------------------
__global__ void dense_gemv(const float* __restrict__ x,
                           const float* __restrict__ W,
                           const float* __restrict__ b,
                           float* __restrict__ out) {
    __shared__ float xs[FLATK];
    for (int k = threadIdx.x; k < FLATK; k += blockDim.x) xs[k] = x[k];
    __syncthreads();

    int n = blockIdx.x * blockDim.x + threadIdx.x;
    if (n >= DOUT) return;
    float acc = b[n];
    for (int k = 0; k < FLATK; ++k) {
        __builtin_prefetch(&W[(k + 8) * DOUT + n], 0, 1);
        acc = fmaf(xs[k], W[k * DOUT + n], acc);
    }
    out[n] = acc > 0.f ? acc : 0.f;
}

// ---------------------------------------------------------------------------
// Box filtering prep: image_pred (7 cols), per-class sort keys (stable-argsort
// semantics via idx tie-break), class-valid mask.
// ---------------------------------------------------------------------------
__global__ void prep_boxes(const float* __restrict__ x,
                           float* __restrict__ image_pred,
                           int* __restrict__ cvmask,
                           float* __restrict__ sortkeys,
                           int* __restrict__ sortvals) {
    int i = blockIdx.x * blockDim.x + threadIdx.x;
    if (i >= NPAD) return;
    if (i < NBOX) {
        const float* p = x + i * VECL;
        float conf = (p[4] > 0.5f) ? 1.0f : 0.0f;
        float pr[VECL];
#pragma unroll
        for (int t = 0; t < VECL; ++t) pr[t] = p[t] * conf;
        int mi = 0;
        float ms = pr[5];
        if (pr[6] > ms) { ms = pr[6]; mi = 1; }
        if (pr[7] > ms) { ms = pr[7]; mi = 2; }
        float ip[7] = {pr[0], pr[1], pr[2], pr[3], pr[4], (float)mi, ms};
#pragma unroll
        for (int t = 0; t < 7; ++t) image_pred[i * 7 + t] = ip[t];
        bool valid = (ip[0] != 0.0f);
#pragma unroll
        for (int c = 0; c < NCLS; ++c) {
            bool cv = valid && (mi == c) && (ms != 0.0f);
            cvmask[c * NBOX + i]   = cv ? 1 : 0;
            sortkeys[c * NPAD + i] = cv ? ip[4] : -__builtin_inff();
            sortvals[c * NPAD + i] = i;
        }
    } else {
#pragma unroll
        for (int c = 0; c < NCLS; ++c) {
            sortkeys[c * NPAD + i] = -__builtin_inff();
            sortvals[c * NPAD + i] = i;
        }
    }
}

// ---------------------------------------------------------------------------
// Bitonic sort, descending by key, ascending idx on ties (== stable argsort
// of -scores). One workgroup per class, data in global scratch.
// ---------------------------------------------------------------------------
static __device__ __forceinline__ bool sort_before(float ka, int va, float kb, int vb) {
    return (ka > kb) || (ka == kb && va < vb);
}

__global__ void bitonic_sort(float* __restrict__ keys, int* __restrict__ vals) {
    float* k = keys + blockIdx.x * NPAD;
    int*   v = vals + blockIdx.x * NPAD;
    for (int size = 2; size <= NPAD; size <<= 1) {
        for (int strd = size >> 1; strd > 0; strd >>= 1) {
            for (int i = threadIdx.x; i < NPAD; i += blockDim.x) {
                int j = i ^ strd;
                if (j > i) {
                    float ki = k[i], kj = k[j];
                    int   vi = v[i], vj = v[j];
                    bool up  = ((i & size) == 0);
                    bool bef = sort_before(ki, vi, kj, vj);
                    if (up ? !bef : bef) {
                        k[i] = kj; k[j] = ki;
                        v[i] = vj; v[j] = vi;
                    }
                }
            }
            __syncthreads();
        }
    }
}

// ---------------------------------------------------------------------------
// Greedy NMS per class (reference semantics). Mask in LDS, boxes in scratch.
// ---------------------------------------------------------------------------
__global__ void nms_kernel(const float* __restrict__ image_pred,
                           const int* __restrict__ cvmask,
                           const int* __restrict__ sortvals,
                           float* __restrict__ sb_ws,
                           float* __restrict__ boxes_out,
                           float* __restrict__ keep_out) {
    int c = blockIdx.x;
    const int* order = sortvals + c * NPAD;
    float* sb = sb_ws + c * NBOX * 7;
    __shared__ int m[NBOX];

    for (int j = threadIdx.x; j < NBOX; j += blockDim.x) {
        int o = order[j];
#pragma unroll
        for (int t = 0; t < 7; ++t) sb[j * 7 + t] = image_pred[o * 7 + t];
        m[j] = cvmask[c * NBOX + o];
    }
    __syncthreads();

    for (int i = 0; i < NBOX; ++i) {
        if (m[i]) {
            float b1x1 = sb[i * 7 + 0], b1y1 = sb[i * 7 + 1];
            float b1x2 = sb[i * 7 + 2], b1y2 = sb[i * 7 + 3];
            float a1 = (b1x2 - b1x1 + 1.0f) * (b1y2 - b1y1 + 1.0f);
            for (int j = threadIdx.x; j < NBOX; j += blockDim.x) {
                if (j > i && m[j]) {
                    float c0 = sb[j * 7 + 0], c1 = sb[j * 7 + 1];
                    float c2 = sb[j * 7 + 2], c3 = sb[j * 7 + 3];
                    float b2x1 = c0 - c2 * 0.5f;
                    float b2y1 = c1 - c3 * 0.5f;
                    float b2x2 = c0, b2y2 = c1;
                    float iw = fminf(fmaxf(fminf(b1x2, b2x2) - fmaxf(b1x1, b2x1) + 1.0f, 0.0f), 2000.0f);
                    float ih = fminf(fmaxf(fminf(b1y2, b2y2) - fmaxf(b1y1, b2y1) + 1.0f, 0.0f), 2000.0f);
                    float inter = iw * ih;
                    float a2 = (b2x2 - b2x1 + 1.0f) * (b2y2 - b2y1 + 1.0f);
                    float iou = inter / (a1 + a2 - inter);
                    if (iou >= 0.4f) m[j] = 0;
                }
            }
        }
        __syncthreads();
    }

    for (int j = threadIdx.x; j < NBOX; j += blockDim.x) {
        int f = m[j];
#pragma unroll
        for (int t = 0; t < 7; ++t)
            boxes_out[(c * NBOX + j) * 7 + t] = f ? sb[j * 7 + t] : 0.0f;
        keep_out[c * NBOX + j] = (float)f;
    }
}

// ---------------------------------------------------------------------------
// Host orchestration
// ---------------------------------------------------------------------------
static inline size_t align256(size_t x) { return (x + 255) & ~size_t(255); }

extern "C" void kernel_launch(void* const* d_in, const int* in_sizes, int n_in,
                              void* d_out, int out_size, void* d_ws, size_t ws_size,
                              hipStream_t stream) {
    (void)in_sizes; (void)n_in; (void)out_size; (void)ws_size;

    const float* img  = (const float*)d_in[0];
    const float* w1   = (const float*)d_in[1];
    const float* b1   = (const float*)d_in[2];
    const float* w2   = (const float*)d_in[3];
    const float* b2   = (const float*)d_in[4];
    const float* w3   = (const float*)d_in[5];
    const float* b3   = (const float*)d_in[6];
    const float* wd   = (const float*)d_in[7];
    const float* bd   = (const float*)d_in[8];
    float* out = (float*)d_out;

    // ---- workspace carve-up -------------------------------------------------
    char* ws = (char*)d_ws;
    size_t off = 0;
    auto alloc = [&](size_t bytes) { char* p = ws + off; off += align256(bytes); return p; };

    const int K1 = 16 * 16 * CIN0, KT1 = K1 / 32, NT1 = C1 / 16;   // 768, 24, 2
    const int K2 = 4 * 4 * C1,     KT2 = K2 / 32, NT2 = C2 / 16;   // 512, 16, 4
    const int K3 = 4 * 4 * C2,     KT3 = K3 / 32, NT3 = C3 / 16;   // 1024, 32, 8

    const size_t B1bytes = (size_t)KT1 * NT1 * 32 * 8 * 4;  //  48 KB
    const size_t B2bytes = (size_t)KT2 * NT2 * 32 * 8 * 4;  //  64 KB
    const size_t B3bytes = (size_t)KT3 * NT3 * 32 * 8 * 4;  // 256 KB (<320KB LDS)

    unsigned int* B1p = (unsigned int*)alloc(B1bytes);
    unsigned int* B2p = (unsigned int*)alloc(B2bytes);
    unsigned int* B3p = (unsigned int*)alloc(B3bytes);

    const int M1 = H1 * W1, M2 = H2 * W2, M3 = H3 * W3;
    float* conv1o = (float*)alloc((size_t)M1 * C1 * 4);
    float* pool1o = (float*)alloc((size_t)HP1 * WP1 * C1 * 4);
    float* conv2o = (float*)alloc((size_t)M2 * C2 * 4);
    float* pool2o = (float*)alloc((size_t)HP2 * WP2 * C2 * 4);
    float* conv3o = (float*)alloc((size_t)FLATK * 4);

    float* image_pred = (float*)alloc((size_t)NBOX * 7 * 4);
    int*   cvmask     = (int*)  alloc((size_t)NCLS * NBOX * 4);
    float* sortkeys   = (float*)alloc((size_t)NCLS * NPAD * 4);
    int*   sortvals   = (int*)  alloc((size_t)NCLS * NPAD * 4);
    float* sb_ws      = (float*)alloc((size_t)NCLS * NBOX * 7 * 4);

    // ---- weight packing -----------------------------------------------------
    pack_b_bf16<<<KT1 * NT1, 32, 0, stream>>>(w1, B1p, K1, C1);
    pack_b_bf16<<<KT2 * NT2, 32, 0, stream>>>(w2, B2p, K2, C2);
    pack_b_bf16<<<KT3 * NT3, 32, 0, stream>>>(w3, B3p, K3, C3);

    // ---- conv1 + pool1 (8 waves/block share B in LDS) -----------------------
    int mt1 = (M1 + 15) / 16;
    conv_wmma_bf16<NT1, CIN0, 16><<<(mt1 + 7) / 8, 256, B1bytes, stream>>>(
        img, B1p, b1, conv1o, M1, W1, WIN, 2, KT1);
    {
        int total = HP1 * WP1 * C1;
        maxpool2<<<(total + 255) / 256, 256, 0, stream>>>(conv1o, pool1o, W1, C1, HP1, WP1);
    }

    // ---- conv2 + pool2 ------------------------------------------------------
    int mt2 = (M2 + 15) / 16;
    conv_wmma_bf16<NT2, C1, 4><<<(mt2 + 7) / 8, 256, B2bytes, stream>>>(
        pool1o, B2p, b2, conv2o, M2, W2, WP1, 4, KT2);
    {
        int total = HP2 * WP2 * C2;
        maxpool2<<<(total + 255) / 256, 256, 0, stream>>>(conv2o, pool2o, W2, C2, HP2, WP2);
    }

    // ---- conv3 (2 M-tiles -> 1 block, 2 waves, full B in LDS) --------------
    conv_wmma_bf16<NT3, C2, 4><<<1, 64, B3bytes, stream>>>(
        pool2o, B3p, b3, conv3o, M3, W3, WP2, 4, KT3);

    // ---- dense (writes x to d_out[0 .. DOUT)) -------------------------------
    dense_gemv<<<DOUT / 256, 256, 0, stream>>>(conv3o, wd, bd, out);

    // ---- box filtering + per-class NMS --------------------------------------
    prep_boxes<<<(NPAD + 255) / 256, 256, 0, stream>>>(out, image_pred, cvmask,
                                                       sortkeys, sortvals);
    bitonic_sort<<<NCLS, 1024, 0, stream>>>(sortkeys, sortvals);
    nms_kernel<<<NCLS, 1024, 0, stream>>>(image_pred, cvmask, sortvals, sb_ws,
                                          out + DOUT,
                                          out + DOUT + NCLS * NBOX * 7);
}
